// Attention_24275155157399
// MI455X (gfx1250) — matrix operand order repaired and checked by reading, not compiled, not run on hardware
//
#include <hip/hip_runtime.h>

// ---------------------------------------------------------------------------
// Grouped-query causal flash attention for gfx1250 (CDNA5, wave32, WMMA).
// B=2, S=2048, H=32, Hkv=8 (G=4), D=128.  fp32 I/O, f16 WMMA math, f32 accum.
// 8 waves x 16 query rows per workgroup; 64-key tiles. K/V staging (the only
// O(S*D) non-matrix work) is amortized across 8 waves. All causal guards are
// scalar so EXEC stays all-ones around WMMA as the ISA requires.
// ---------------------------------------------------------------------------

typedef __attribute__((ext_vector_type(16))) _Float16 v16h;
typedef __attribute__((ext_vector_type(8)))  _Float16 v8h;
typedef __attribute__((ext_vector_type(4)))  _Float16 v4h;
typedef __attribute__((ext_vector_type(8)))  float    v8f;

union AF16 { v16h v; v8h h2[2]; _Float16 e[16]; };

#define WMMA_F16F32(A, Bm, C) \
  __builtin_amdgcn_wmma_f32_16x16x32_f16(false, (A), false, (Bm), (short)0, (C), false, false)

#define V8F_ZERO ((v8f){0.f,0.f,0.f,0.f,0.f,0.f,0.f,0.f})

__global__ __launch_bounds__(256) void gqa_causal_fa_kernel(
    const float* __restrict__ q, const float* __restrict__ k,
    const float* __restrict__ v, float* __restrict__ out)
{
    constexpr int S    = 2048;
    constexpr int H    = 32;
    constexpr int HKV  = 8;
    constexpr int D    = 128;
    constexpr int G    = 4;      // H / HKV
    constexpr int QBLK = 128;    // queries per workgroup (8 waves x 16 rows)
    constexpr int KBLK = 64;     // keys per iteration (4 x 16-wide sub-tiles)
    constexpr float scale = 0.08838834764831845f; // D^-0.5

    __shared__ __align__(32) _Float16 sK [KBLK][D];      // 16 KB keys row-major
    __shared__ __align__(32) _Float16 sVt[D][KBLK];      // 16 KB values transposed
    __shared__ __align__(32) _Float16 sP [8][16][KBLK];  // 16 KB probs per wave

    const int nqb = S / QBLK;                   // 16
    const int qb  = blockIdx.x % nqb;
    const int h   = (blockIdx.x / nqb) % H;
    const int b   = blockIdx.x / (nqb * H);
    const int kvh = h / G;

    const int tid  = threadIdx.x;
    // wave id is wave-uniform: force it into an SGPR so causal guards become
    // scalar branches (EXEC must stay all-ones around WMMA).
    const int w    = __builtin_amdgcn_readfirstlane(tid >> 5);   // 0..7
    const int lane = tid & 31;
    const int lm   = lane & 15;                 // N / M sub-index
    const int hi8  = (lane >> 4) * 8;           // C-row offset, upper half-wave
    const int hi16 = (lane >> 4) * 16;          // B K-offset, upper half-wave

    const int q0   = qb * QBLK + w * 16;        // first query row (scalar)
    const int q_hi = q0 + 15;                   // last query row (scalar)

    // ---- Q tile (16 x 128) -> 4 A-fragments, pre-scaled by D^-0.5 ----------
    v16h aq[4];
    {
        const float* qp = q + (((size_t)b * S + (size_t)(q0 + lm)) * H + h) * D;
        #pragma unroll
        for (int ks = 0; ks < 4; ++ks) {
            const float* p0 = qp + ks * 32 + hi16;       // halves 0..7
            const float* p1 = qp + ks * 32 + hi16 + 8;   // halves 8..15
            float4 f0 = *(const float4*)(p0);
            float4 f1 = *(const float4*)(p0 + 4);
            float4 f2 = *(const float4*)(p1);
            float4 f3 = *(const float4*)(p1 + 4);
            AF16 u;
            u.e[0]=(_Float16)(f0.x*scale); u.e[1]=(_Float16)(f0.y*scale);
            u.e[2]=(_Float16)(f0.z*scale); u.e[3]=(_Float16)(f0.w*scale);
            u.e[4]=(_Float16)(f1.x*scale); u.e[5]=(_Float16)(f1.y*scale);
            u.e[6]=(_Float16)(f1.z*scale); u.e[7]=(_Float16)(f1.w*scale);
            u.e[8]=(_Float16)(f2.x*scale); u.e[9]=(_Float16)(f2.y*scale);
            u.e[10]=(_Float16)(f2.z*scale); u.e[11]=(_Float16)(f2.w*scale);
            u.e[12]=(_Float16)(f3.x*scale); u.e[13]=(_Float16)(f3.y*scale);
            u.e[14]=(_Float16)(f3.z*scale); u.e[15]=(_Float16)(f3.w*scale);
            aq[ks] = u.v;
        }
    }

    // ---- flash-attention state ---------------------------------------------
    float m_i[8], l_i[8];
    v8f   oacc[8];                       // O accumulators: 8 d-slices x 8 rows
    #pragma unroll
    for (int r = 0; r < 8; ++r) { m_i[r] = -1.0e30f; l_i[r] = 0.0f; }
    #pragma unroll
    for (int t = 0; t < 8; ++t) oacc[t] = V8F_ZERO;

    // causal: workgroup needs keys up to qb*128+127 -> (2*qb + 2) tiles of 64
    const int nkb = 2 * qb + 2;

    for (int jb = 0; jb < nkb; ++jb) {
        // ---- cooperative K/V tile staging: 64 keys x 128 d, f32 -> f16 -----
        __syncthreads();                 // previous tile fully consumed
        const int lrow = tid >> 2;               // 0..63 (K staging row)
        const int lcb  = (tid & 3) * 32;         // 32-wide column chunk
        {
            // K: row-major, packed v4h stores (ds_store_b64)
            const float* kp =
                k + (((size_t)b * S + jb * KBLK + lrow) * HKV + kvh) * D + lcb;
            #pragma unroll
            for (int i = 0; i < 8; ++i) {
                float4 kf = *(const float4*)(kp + i * 4);
                v4h pk = {(_Float16)kf.x, (_Float16)kf.y,
                          (_Float16)kf.z, (_Float16)kf.w};
                *(v4h*)&sK[lrow][lcb + i * 4] = pk;
            }
            // V: transpose via 4x4 register blocks (coalesced b128 loads,
            // packed v4h stores along the key dimension). Wave w owns keys
            // w*8 .. w*8+7; each lane owns 4 consecutive d columns.
            const int kb0 = w * 8;
            const int d0  = lane * 4;
            const float* vp =
                v + (((size_t)b * S + jb * KBLK + kb0) * HKV + kvh) * D + d0;
            #pragma unroll
            for (int j = 0; j < 2; ++j) {
                const float* vpj = vp + (size_t)(j * 4) * (HKV * D);
                float4 r0 = *(const float4*)(vpj + 0 * (HKV * D));
                float4 r1 = *(const float4*)(vpj + 1 * (HKV * D));
                float4 r2 = *(const float4*)(vpj + 2 * (HKV * D));
                float4 r3 = *(const float4*)(vpj + 3 * (HKV * D));
                const int kk = kb0 + j * 4;
                v4h c0 = {(_Float16)r0.x,(_Float16)r1.x,(_Float16)r2.x,(_Float16)r3.x};
                v4h c1 = {(_Float16)r0.y,(_Float16)r1.y,(_Float16)r2.y,(_Float16)r3.y};
                v4h c2 = {(_Float16)r0.z,(_Float16)r1.z,(_Float16)r2.z,(_Float16)r3.z};
                v4h c3 = {(_Float16)r0.w,(_Float16)r1.w,(_Float16)r2.w,(_Float16)r3.w};
                *(v4h*)&sVt[d0 + 0][kk] = c0;
                *(v4h*)&sVt[d0 + 1][kk] = c1;
                *(v4h*)&sVt[d0 + 2][kk] = c2;
                *(v4h*)&sVt[d0 + 3][kk] = c3;
            }
        }
        __syncthreads();

        // ---- warm L2/L0 for the next tile while we compute this one --------
        if (jb + 1 < nkb) {
            const size_t off =
                (((size_t)b * S + (jb + 1) * KBLK + lrow) * HKV + kvh) * D + lcb;
            __builtin_prefetch(k + off, 0, 3);
            __builtin_prefetch(v + off, 0, 3);
        }

        const int  kbase = jb * KBLK;             // scalar
        const bool act0  = kbase        <= q_hi;  // scalar: tile intersects rows
        const bool act2  = (kbase + 32) <= q_hi;  // scalar: upper 32 keys needed

        if (act0) {
            // ---- S = Q * K^T : four 16x16 f32 tiles (inline-0 first MAC) ---
            v8f sc[4];
            #pragma unroll
            for (int nt = 0; nt < 2; ++nt) {
                v16h bk = *(const v16h*)&sK[nt * 16 + lm][hi16];
                sc[nt] = WMMA_F16F32(aq[0], bk, V8F_ZERO);
                #pragma unroll
                for (int ks = 1; ks < 4; ++ks) {
                    bk = *(const v16h*)&sK[nt * 16 + lm][ks * 32 + hi16];
                    sc[nt] = WMMA_F16F32(aq[ks], bk, sc[nt]);
                }
            }
            if (act2) {
                #pragma unroll
                for (int nt = 2; nt < 4; ++nt) {
                    v16h bk = *(const v16h*)&sK[nt * 16 + lm][hi16];
                    sc[nt] = WMMA_F16F32(aq[0], bk, V8F_ZERO);
                    #pragma unroll
                    for (int ks = 1; ks < 4; ++ks) {
                        bk = *(const v16h*)&sK[nt * 16 + lm][ks * 32 + hi16];
                        sc[nt] = WMMA_F16F32(aq[ks], bk, sc[nt]);
                    }
                }
            } else {
                sc[2] = V8F_ZERO;
                sc[3] = V8F_ZERO;
            }

            // ---- causal mask + online softmax, emit P(f16) to LDS scratch --
            #pragma unroll
            for (int r = 0; r < 8; ++r) {
                const int srow = q0 + r + hi8;  // query row for this register
                float x[4];
                #pragma unroll
                for (int nt = 0; nt < 4; ++nt) {
                    float xv = sc[nt][r];
                    if (kbase + nt * 16 + lm > srow) xv = -1.0e30f;
                    x[nt] = xv;
                }
                float mx = fmaxf(fmaxf(x[0], x[1]), fmaxf(x[2], x[3]));
                mx = fmaxf(mx, __shfl_xor(mx, 1));
                mx = fmaxf(mx, __shfl_xor(mx, 2));
                mx = fmaxf(mx, __shfl_xor(mx, 4));
                mx = fmaxf(mx, __shfl_xor(mx, 8));
                const float mnew  = fmaxf(m_i[r], mx);
                const float alpha = __expf(m_i[r] - mnew);
                m_i[r] = mnew;
                float p[4];
                #pragma unroll
                for (int nt = 0; nt < 4; ++nt) p[nt] = __expf(x[nt] - mnew);
                float rs = (p[0] + p[1]) + (p[2] + p[3]);
                rs += __shfl_xor(rs, 1);
                rs += __shfl_xor(rs, 2);
                rs += __shfl_xor(rs, 4);
                rs += __shfl_xor(rs, 8);
                l_i[r] = l_i[r] * alpha + rs;
                #pragma unroll
                for (int t = 0; t < 8; ++t) oacc[t][r] *= alpha;
                sP[w][r + hi8][lm]      = (_Float16)p[0];
                sP[w][r + hi8][16 + lm] = (_Float16)p[1];
                if (act2) {
                    sP[w][r + hi8][32 + lm] = (_Float16)p[2];
                    sP[w][r + hi8][48 + lm] = (_Float16)p[3];
                }
            }
            // within-wave LDS store -> load fence for the P transpose
            asm volatile("s_wait_dscnt 0" ::: "memory");

            // ---- O += P * V : A-frags from sP, B-frags from transposed V ---
            AF16 ap0;
            ap0.h2[0] = *(const v8h*)&sP[w][lm][hi16];       // keys  0..31
            ap0.h2[1] = *(const v8h*)&sP[w][lm][hi16 + 8];
            #pragma unroll
            for (int ds = 0; ds < 8; ++ds) {
                v16h bv = *(const v16h*)&sVt[ds * 16 + lm][hi16];
                oacc[ds] = WMMA_F16F32(ap0.v, bv, oacc[ds]);
            }
            if (act2) {                                      // keys 32..63
                AF16 ap1;
                ap1.h2[0] = *(const v8h*)&sP[w][lm][32 + hi16];
                ap1.h2[1] = *(const v8h*)&sP[w][lm][40 + hi16];
                #pragma unroll
                for (int ds = 0; ds < 8; ++ds) {
                    v16h bv = *(const v16h*)&sVt[ds * 16 + lm][32 + hi16];
                    oacc[ds] = WMMA_F16F32(ap1.v, bv, oacc[ds]);
                }
            }
        }
    }

    // ---- normalize and write O (coalesced over d within 16-lane groups) ----
    float inv[8];
    #pragma unroll
    for (int r = 0; r < 8; ++r) inv[r] = 1.0f / l_i[r];
    #pragma unroll
    for (int ds = 0; ds < 8; ++ds) {
        #pragma unroll
        for (int r = 0; r < 8; ++r) {
            const int srow = q0 + r + hi8;
            out[(((size_t)b * S + srow) * H + h) * D + ds * 16 + lm] =
                oacc[ds][r] * inv[r];
        }
    }
}

extern "C" void kernel_launch(void* const* d_in, const int* in_sizes, int n_in,
                              void* d_out, int out_size, void* d_ws, size_t ws_size,
                              hipStream_t stream) {
    (void)in_sizes; (void)n_in; (void)out_size; (void)d_ws; (void)ws_size;
    const float* q = (const float*)d_in[0];
    const float* k = (const float*)d_in[1];
    const float* v = (const float*)d_in[2];
    float* out = (float*)d_out;

    constexpr int B = 2, S = 2048, H = 32, QBLK = 128;
    dim3 grid(B * H * (S / QBLK));   // 1024 workgroups
    dim3 block(256);                 // 8 waves (wave32)
    gqa_causal_fa_kernel<<<grid, block, 0, stream>>>(q, k, v, out);
}
